// CodingLoss_65798898975305
// MI455X (gfx1250) — matrix operand order repaired and checked by reading, not compiled, not run on hardware
//
#include <hip/hip_runtime.h>
#include <hip/hip_bf16.h>
#include <math.h>

#define B_ROWS 16384
#define NCLS   2048
#define DLEN   2048

#define BM 64      // rows per block
#define BN 128     // class columns per n-tile
#define KT 32      // K per stage step (one bf16 WMMA K)
#define LDP 40     // padded LDS row stride (bf16): 64B data + 16B pad -> conflict-free b128 frags

typedef __bf16 bf16_t;
typedef __bf16 v16bf __attribute__((ext_vector_type(16)));
typedef __bf16 v8bf  __attribute__((ext_vector_type(8)));
typedef float  v8f   __attribute__((ext_vector_type(8)));
typedef float  v4f   __attribute__((ext_vector_type(4)));
typedef unsigned int v4u __attribute__((ext_vector_type(4)));
typedef int    v8i   __attribute__((ext_vector_type(8)));
typedef int    v4i   __attribute__((ext_vector_type(4)));

// ---------------- row sums of X: sx[b] = sum_d x[b,d] ----------------
__global__ void rowsum_x_kernel(const float* __restrict__ x, float* __restrict__ sx) {
    const int wave = threadIdx.x >> 5;
    const int lane = threadIdx.x & 31;
    const int row  = blockIdx.x * 8 + wave;
    const float* p = x + (size_t)row * DLEN;
    float s = 0.f;
    #pragma unroll 8
    for (int j = lane; j < DLEN; j += 32) s += p[j];
    #pragma unroll
    for (int m = 16; m >= 1; m >>= 1) s += __shfl_xor(s, m, 32);
    if (lane == 0) sx[row] = s;
}

// ------- codebook pass: sc[c] = sum_d cb[c,d]; split cb into bf16 hi/lo -------
__global__ void prep_codebook_kernel(const float* __restrict__ cb, float* __restrict__ sc,
                                     bf16_t* __restrict__ ch, bf16_t* __restrict__ cl) {
    const int c = blockIdx.x;
    const float* p = cb + (size_t)c * DLEN;
    float s = 0.f;
    for (int j = threadIdx.x; j < DLEN; j += 256) {
        float v = p[j];
        s += v;
        bf16_t hi = (bf16_t)v;
        float  lf = v - (float)hi;
        ch[(size_t)c * DLEN + j] = hi;
        cl[(size_t)c * DLEN + j] = (bf16_t)lf;
    }
    __shared__ float red[8];
    #pragma unroll
    for (int m = 16; m >= 1; m >>= 1) s += __shfl_xor(s, m, 32);
    if ((threadIdx.x & 31) == 0) red[threadIdx.x >> 5] = s;
    __syncthreads();
    if (threadIdx.x == 0) {
        float t = 0.f;
        #pragma unroll
        for (int i = 0; i < 8; ++i) t += red[i];
        sc[c] = t;
    }
}

// ---- TDM: load a BN x KT bf16 tile (row stride DLEN elems) into padded LDS ----
// D# per ISA ch.8: 2D tile, data_size=2B, pad every 16 DWORDs (64B row) by 4 DWORDs (16B)
__device__ __forceinline__ void tdm_load_tile(const bf16_t* gsrc, unsigned lds_addr) {
    unsigned long long ga = (unsigned long long)(size_t)gsrc;
    v4u g0;
    g0[0] = 1u;                                     // count=1, user descriptor
    g0[1] = lds_addr;                               // LDS byte address
    g0[2] = (unsigned)ga;                           // global_addr[31:0]
    g0[3] = ((unsigned)(ga >> 32) & 0x01FFFFFFu) | (2u << 30);  // global_addr[56:32] | type=2
    v8i g1;
    g1[0] = (1 << 16)            // data_size = 2 bytes
          | (1 << 20)            // pad_enable
          | (3 << 22)            // pad_interval: 16 DWORDs of data
          | (3 << 25);           // pad_amount:   4 DWORDs of pad
    g1[1] = (int)(((unsigned)DLEN & 0xFFFFu) << 16);                       // tensor_dim0 lo
    g1[2] = (int)((((unsigned)DLEN >> 16) & 0xFFFFu) | (((unsigned)NCLS & 0xFFFFu) << 16)); // td0 hi | td1 lo
    g1[3] = (int)((((unsigned)NCLS >> 16) & 0xFFFFu) | ((unsigned)KT << 16));               // td1 hi | tile_dim0
    g1[4] = BN;                  // tile_dim1 = 128 rows, tile_dim2 = 0
    g1[5] = DLEN;                // tensor_dim0_stride[31:0] (elements)
    g1[6] = 0;
    g1[7] = 0;
#if __clang_major__ >= 23
    __builtin_amdgcn_tensor_load_to_lds(g0, g1, (v4i)0, (v4i)0, (v8i)0, 0);
#else
    __builtin_amdgcn_tensor_load_to_lds(g0, g1, (v4i)0, (v4i)0, 0);
#endif
}

// ---------------- main fused GEMM + online logsumexp ----------------
// y[b,c] = -dist[b,c] = 2*S + D - sx[b] - sc[c];  loss_b = logsumexp_c(y) - y[b,lab_b]
__global__ void __launch_bounds__(256)
coding_loss_main_kernel(const float* __restrict__ x, const int* __restrict__ labels,
                        const bf16_t* __restrict__ ch, const bf16_t* __restrict__ cl,
                        const float* __restrict__ sx, const float* __restrict__ sc,
                        float* __restrict__ blockSums) {
    __shared__ bf16_t Xh[2][BM][LDP], Xl[2][BM][LDP];
    __shared__ bf16_t Bh[2][BN][LDP], Bl[2][BN][LDP];
    __shared__ float  sxs[BM];
    __shared__ int    labs[BM];
    __shared__ float  ylab[BM];
    __shared__ float  pm[4][BM], ps[4][BM];
    __shared__ float  lossArr[BM];

    const int tid   = threadIdx.x;
    const int lane  = tid & 31;
    const int wid   = tid >> 5;
    const int wm    = wid >> 2;     // 0..1 : 32-row strip
    const int wn    = wid & 3;      // 0..3 : 32-col strip within n-tile
    const int col16 = lane & 15;
    const int half  = lane >> 4;
    const int r0    = blockIdx.x * BM;

    const int sxrow = tid >> 2;          // X staging: row 0..63
    const int sxko  = (tid & 3) * 8;     // X staging: k offset 0/8/16/24

    if (tid < BM) { sxs[tid] = sx[r0 + tid]; labs[tid] = labels[r0 + tid]; }

    // X-stage helpers: load global f32, split to bf16 hi/lo, store to LDS
    auto load_x = [&](int k0, v4f& a, v4f& b) {
        const float* gp = x + (size_t)(r0 + sxrow) * DLEN + k0 + sxko;
        a = *(const v4f*)gp;
        b = *(const v4f*)(gp + 4);
    };
    auto store_x = [&](int buf, v4f a, v4f b) {
        v8bf hi, lo;
        #pragma unroll
        for (int j = 0; j < 4; ++j) {
            float v = a[j]; bf16_t h = (bf16_t)v;
            hi[j] = h; lo[j] = (bf16_t)(v - (float)h);
        }
        #pragma unroll
        for (int j = 0; j < 4; ++j) {
            float v = b[j]; bf16_t h = (bf16_t)v;
            hi[4 + j] = h; lo[4 + j] = (bf16_t)(v - (float)h);
        }
        *(v8bf*)&Xh[buf][sxrow][sxko] = hi;
        *(v8bf*)&Xl[buf][sxrow][sxko] = lo;
    };
    auto issue_b = [&](int buf, int n0, int k0) {
        if (wid == 0) {
            const size_t g = (size_t)n0 * DLEN + k0;
            tdm_load_tile(ch + g, (unsigned)(size_t)&Bh[buf][0][0]);
            tdm_load_tile(cl + g, (unsigned)(size_t)&Bl[buf][0][0]);
        }
    };

    // online logsumexp state, per lane for rows (wm*32 + mi*16 + 8*half + i)
    float rm[2][8], rs[2][8];
    #pragma unroll
    for (int mi = 0; mi < 2; ++mi)
        #pragma unroll
        for (int i = 0; i < 8; ++i) { rm[mi][i] = -__builtin_inff(); rs[mi][i] = 0.f; }

    for (int nt = 0; nt < NCLS / BN; ++nt) {
        const int n0 = nt * BN;
        v8f acc[2][2];
        #pragma unroll
        for (int mi = 0; mi < 2; ++mi)
            #pragma unroll
            for (int ni = 0; ni < 2; ++ni) acc[mi][ni] = (v8f)0.0f;

        // ---- prologue: fill buffer 0 ----
        {
            v4f a, b;
            load_x(0, a, b);
            issue_b(0, n0, 0);
            store_x(0, a, b);
            if (wid == 0) __builtin_amdgcn_s_wait_tensorcnt(0);
            __syncthreads();
        }

        for (int kt = 0; kt < DLEN / KT; ++kt) {
            const int cur = kt & 1, nxt = cur ^ 1;
            const bool have_next = (kt + 1) < (DLEN / KT);
            v4f a, b;
            if (have_next) {
                load_x((kt + 1) * KT, a, b);        // global loads overlap WMMAs
                issue_b(nxt, n0, (kt + 1) * KT);    // TDM overlaps WMMAs
            }

            // ---- load fragments from current buffers ----
            // A 16x32 bf16: elem j -> K = half*8 + j ; elem 8+j -> K = 16 + half*8 + j
            v16bf ah[2], al[2], bh[2], bl[2];
            #pragma unroll
            for (int mi = 0; mi < 2; ++mi) {
                const int r = wm * 32 + mi * 16 + col16;
                v8bf c0 = *(const v8bf*)(&Xh[cur][r][0] + half * 8);
                v8bf c1 = *(const v8bf*)(&Xh[cur][r][0] + 16 + half * 8);
                v8bf d0 = *(const v8bf*)(&Xl[cur][r][0] + half * 8);
                v8bf d1 = *(const v8bf*)(&Xl[cur][r][0] + 16 + half * 8);
                #pragma unroll
                for (int j = 0; j < 8; ++j) {
                    ah[mi][j] = c0[j]; ah[mi][8 + j] = c1[j];
                    al[mi][j] = d0[j]; al[mi][8 + j] = d1[j];
                }
            }
            // B 32x16 bf16: lane = col N; elem j -> K = half*8 + j ; elem 8+j -> K = 16 + half*8 + j
            #pragma unroll
            for (int ni = 0; ni < 2; ++ni) {
                const int r = wn * 32 + ni * 16 + col16;
                v8bf c0 = *(const v8bf*)(&Bh[cur][r][0] + half * 8);
                v8bf c1 = *(const v8bf*)(&Bh[cur][r][0] + 16 + half * 8);
                v8bf d0 = *(const v8bf*)(&Bl[cur][r][0] + half * 8);
                v8bf d1 = *(const v8bf*)(&Bl[cur][r][0] + 16 + half * 8);
                #pragma unroll
                for (int j = 0; j < 8; ++j) {
                    bh[ni][j] = c0[j]; bh[ni][8 + j] = c1[j];
                    bl[ni][j] = d0[j]; bl[ni][8 + j] = d1[j];
                }
            }
            // ---- bf16x3 split-precision WMMA: hi*hi + hi*lo + lo*hi ----
            #pragma unroll
            for (int mi = 0; mi < 2; ++mi)
                #pragma unroll
                for (int ni = 0; ni < 2; ++ni) {
                    acc[mi][ni] = __builtin_amdgcn_wmma_f32_16x16x32_bf16(
                        false, ah[mi], false, bh[ni], (short)0, acc[mi][ni], false, false);
                    acc[mi][ni] = __builtin_amdgcn_wmma_f32_16x16x32_bf16(
                        false, ah[mi], false, bl[ni], (short)0, acc[mi][ni], false, false);
                    acc[mi][ni] = __builtin_amdgcn_wmma_f32_16x16x32_bf16(
                        false, al[mi], false, bh[ni], (short)0, acc[mi][ni], false, false);
                }

            if (have_next) store_x(nxt, a, b);
            if (wid == 0) __builtin_amdgcn_s_wait_tensorcnt(0);
            __syncthreads();
        }

        // ---- epilogue for this n-tile: y = 2S + D - sx - sc, online logsumexp ----
        #pragma unroll
        for (int mi = 0; mi < 2; ++mi)
            #pragma unroll
            for (int ni = 0; ni < 2; ++ni) {
                const int colg = n0 + wn * 32 + ni * 16 + col16;
                const float scc = sc[colg];
                float yv[8];
                #pragma unroll
                for (int i = 0; i < 8; ++i) {
                    const int rl = wm * 32 + mi * 16 + 8 * half + i;
                    float y = 2.f * acc[mi][ni][i] + (float)DLEN - sxs[rl] - scc;
                    yv[i] = y;
                    if (labs[rl] == colg) ylab[rl] = y;   // exactly one writer per row
                }
                #pragma unroll
                for (int i = 0; i < 8; ++i) {
                    float m = yv[i];
                    #pragma unroll
                    for (int s = 1; s < 16; s <<= 1) m = fmaxf(m, __shfl_xor(m, s, 32));
                    float e = expf(yv[i] - m);
                    #pragma unroll
                    for (int s = 1; s < 16; s <<= 1) e += __shfl_xor(e, s, 32);
                    const float om = rm[mi][i];
                    const float nm = fmaxf(om, m);
                    rs[mi][i] = rs[mi][i] * expf(om - nm) + e * expf(m - nm);
                    rm[mi][i] = nm;
                }
            }
    }

    // ---- cross-wave merge (4 column-waves share each row) ----
    if (col16 == 0) {
        #pragma unroll
        for (int mi = 0; mi < 2; ++mi)
            #pragma unroll
            for (int i = 0; i < 8; ++i) {
                const int rl = wm * 32 + mi * 16 + 8 * half + i;
                pm[wn][rl] = rm[mi][i];
                ps[wn][rl] = rs[mi][i];
            }
    }
    __syncthreads();
    if (tid < BM) {
        float M = pm[0][tid];
        #pragma unroll
        for (int w = 1; w < 4; ++w) M = fmaxf(M, pm[w][tid]);
        float S = 0.f;
        #pragma unroll
        for (int w = 0; w < 4; ++w) S += ps[w][tid] * expf(pm[w][tid] - M);
        lossArr[tid] = M + logf(S) - ylab[tid];
    }
    __syncthreads();
    if (tid == 0) {
        float t = 0.f;
        for (int i = 0; i < BM; ++i) t += lossArr[i];
        blockSums[blockIdx.x] = t;
    }
}

// ---------------- final deterministic mean ----------------
__global__ void finalize_kernel(const float* __restrict__ bs, int n, float* __restrict__ out) {
    float s = 0.f;
    for (int i = threadIdx.x; i < n; i += 256) s += bs[i];
    #pragma unroll
    for (int m = 16; m >= 1; m >>= 1) s += __shfl_xor(s, m, 32);
    __shared__ float red[8];
    if ((threadIdx.x & 31) == 0) red[threadIdx.x >> 5] = s;
    __syncthreads();
    if (threadIdx.x == 0) {
        float t = 0.f;
        #pragma unroll
        for (int i = 0; i < 8; ++i) t += red[i];
        out[0] = t / (float)B_ROWS;
    }
}

extern "C" void kernel_launch(void* const* d_in, const int* in_sizes, int n_in,
                              void* d_out, int out_size, void* d_ws, size_t ws_size,
                              hipStream_t stream) {
    const float* x      = (const float*)d_in[0];
    const int*   labels = (const int*)d_in[1];
    const float* cb     = (const float*)d_in[2];

    char*   w     = (char*)d_ws;
    float*  sc    = (float*)w;                                   //   8 KB
    float*  sx    = (float*)(w + 8 * 1024);                      //  64 KB
    float*  bsums = (float*)(w + 72 * 1024);                     //   1 KB
    bf16_t* chp   = (bf16_t*)(w + 128 * 1024);                   //   8 MB
    bf16_t* clp   = (bf16_t*)(w + 128 * 1024 + (size_t)NCLS * DLEN * sizeof(bf16_t)); // 8 MB

    prep_codebook_kernel<<<NCLS, 256, 0, stream>>>(cb, sc, chp, clp);
    rowsum_x_kernel<<<B_ROWS / 8, 256, 0, stream>>>(x, sx);
    coding_loss_main_kernel<<<B_ROWS / BM, 256, 0, stream>>>(x, labels, chp, clp, sx, sc, bsums);
    finalize_kernel<<<1, 256, 0, stream>>>(bsums, B_ROWS / BM, (float*)d_out);
}
